// RSSMRollout_36833639530595
// MI455X (gfx1250) — compile-verified, matrix operand order checked by reading, not run on hardware
//
#include <hip/hip_runtime.h>
#include <hip/hip_bf16.h>

typedef __attribute__((ext_vector_type(16))) __bf16 bf16x16;
typedef __attribute__((ext_vector_type(8)))  float  f32x8;

#define TT 64
#define BB 1024
#define EE 1024
#define AA 6
#define SS 30
#define DD 200
#define HH 200
#define HP 208   // H/D padded to 13 n-tiles of 16
#define KD 224   // K=200 padded to 7 k-tiles of 32

// ---- workspace layout (bytes). One WMMA B-tile = 32 lanes * 16 bf16 = 1024 B.
static constexpr size_t TILE_B   = 1024;
static constexpr size_t OFF_WRI  = 0;                        // 1x13 tiles (stoch part of Wri)
static constexpr size_t OFF_WIHR = OFF_WRI  + 13  * TILE_B;  // 7x13 each
static constexpr size_t OFF_WIHZ = OFF_WIHR + 91  * TILE_B;
static constexpr size_t OFF_WIHN = OFF_WIHZ + 91  * TILE_B;
static constexpr size_t OFF_WHHR = OFF_WIHN + 91  * TILE_B;
static constexpr size_t OFF_WHHZ = OFF_WHHR + 91  * TILE_B;
static constexpr size_t OFF_WHHN = OFF_WHHZ + 91  * TILE_B;
static constexpr size_t OFF_WP1  = OFF_WHHN + 91  * TILE_B;
static constexpr size_t OFF_WP2  = OFF_WP1  + 91  * TILE_B;  // 7x4
static constexpr size_t OFF_WQ1D = OFF_WP2  + 28  * TILE_B;
static constexpr size_t OFF_WQ1E = OFF_WQ1D + 91  * TILE_B;  // 32x13
static constexpr size_t OFF_WQ2  = OFF_WQ1E + 416 * TILE_B;  // 7x4
static constexpr size_t OFF_EMBQ = OFF_WQ2  + 28  * TILE_B;  // fp32 [T*B][HP]
static constexpr size_t OFF_ACTRI= OFF_EMBQ + (size_t)TT*BB*HP*4;

// ---------------------------------------------------------------- helpers
__device__ __forceinline__ unsigned short f2bf(float f) {
    unsigned u = __float_as_uint(f);
    u += 0x7fffu + ((u >> 16) & 1u);          // round-to-nearest-even
    return (unsigned short)(u >> 16);
}
__device__ __forceinline__ float elu1(float x)  { return x > 0.f ? x : (__expf(x) - 1.f); }
__device__ __forceinline__ float sigm(float x)  { return 1.f / (1.f + __expf(-x)); }
__device__ __forceinline__ float softp01(float x){ return (x > 20.f ? x : log1pf(__expf(x))) + 0.1f; }

union Frag { bf16x16 v; unsigned u[8]; };

// A fragment (16x32 bf16, ISA 7.12.2): lane m = L&15, group g = L>>4;
// VGPR v<4 -> K pair 2v + 8g, v>=4 -> 16 + 2(v-4) + 8g.  Source: row-major bf16 LDS.
__device__ __forceinline__ void loadA(const unsigned short* p, int stride, int m, int g,
                                      int k0, Frag& a) {
    const int base = m * stride + k0 + g * 8;
#pragma unroll
    for (int v = 0; v < 8; ++v) {
        int kp = (v < 4) ? (2 * v) : (16 + 2 * (v - 4));
        a.u[v] = *(const unsigned*)(p + base + kp);
    }
}
// B fragment: pre-packed per-lane-contiguous, lane L reads 32 B (two b128 loads).
__device__ __forceinline__ void loadB(const unsigned* wbase, int tile, int L, Frag& b) {
    const unsigned* p = wbase + (size_t)tile * 256 + L * 8;
#pragma unroll
    for (int v = 0; v < 8; ++v) b.u[v] = p[v];
}
__device__ __forceinline__ f32x8 wmma_bf16(const Frag& a, const Frag& b, f32x8 c) {
    return __builtin_amdgcn_wmma_f32_16x16x32_bf16(false, a.v, false, b.v, (short)0, c,
                                                   false, false);
}

// ------------------------------------------------- weight pack: fp32 -> bf16 fragment tiles
__global__ __launch_bounds__(32) void k_pack(const float* __restrict__ src, int ld, int row0,
                                             int col0, int Ks, int Ns, int KT,
                                             unsigned short* __restrict__ dst) {
    int kt = blockIdx.x, nt = blockIdx.y, L = threadIdx.x;
    int g = L >> 4, n = L & 15;
    unsigned short* out = dst + ((size_t)(nt * KT + kt)) * 512 + L * 16;
#pragma unroll
    for (int v = 0; v < 8; ++v)
#pragma unroll
        for (int h = 0; h < 2; ++h) {
            int kl = ((v < 4) ? (2 * v) : (16 + 2 * (v - 4))) + g * 8 + h;
            int k = kt * 32 + kl;
            int nn = nt * 16 + n;
            float val = (k < Ks && nn < Ns) ? src[(size_t)(row0 + k) * ld + col0 + nn] : 0.f;
            out[v * 2 + h] = f2bf(val);
        }
}

// ------------------------------------------------- actri = action @ Wri[:6,:] + bri  (K=6, scalar)
__global__ void k_actri(const float* __restrict__ act, const float* __restrict__ Wri,
                        const float* __restrict__ bri, float* __restrict__ out) {
    long long idx = (long long)blockIdx.x * blockDim.x + threadIdx.x;
    long long total = (long long)TT * BB * HP;
    if (idx >= total) return;
    int h = (int)(idx % HP);
    long long row = idx / HP;
    float v = 0.f;
    if (h < HH) {
        v = bri[h];
#pragma unroll
        for (int a = 0; a < AA; ++a) v += act[row * AA + a] * Wri[a * HH + h];
    }
    out[row * HP + h] = v;
}

// ------------------------------------------------- embq = obs_embed @ Wq1[D:,:] + bq1
// Each block: one 16-row M tile x TWO 16-col N tiles (amortize LDS staging over 2 WMMAs).
__global__ __launch_bounds__(32) void k_embq(const float* __restrict__ obs,
                                             const unsigned* __restrict__ wq1e,
                                             const float* __restrict__ bq1,
                                             float* __restrict__ embq) {
    __shared__ unsigned short tile[16 * 32];
    int mt = blockIdx.x, L = threadIdx.x;
    int j0 = blockIdx.y * 2, j1 = j0 + 1;
    bool has1 = (j1 < 13);
    int g = L >> 4, lm = L & 15;
    f32x8 acc0 = {0.f, 0.f, 0.f, 0.f, 0.f, 0.f, 0.f, 0.f};
    f32x8 acc1 = acc0;
    for (int kt = 0; kt < 32; ++kt) {
#pragma unroll
        for (int r = 0; r < 16; ++r)
            tile[r * 32 + L] = f2bf(obs[((size_t)(mt * 16 + r)) * EE + kt * 32 + L]);
        __syncthreads();
        Frag a, b0, b1;
        loadA(tile, 32, lm, g, 0, a);
        loadB(wq1e, j0 * 32 + kt, L, b0);
        acc0 = wmma_bf16(a, b0, acc0);
        if (has1) {
            loadB(wq1e, j1 * 32 + kt, L, b1);
            acc1 = wmma_bf16(a, b1, acc1);
        }
        __syncthreads();
    }
    // column-invariant bias hoisted out of the per-row loop
    int col0 = j0 * 16 + lm, col1 = j1 * 16 + lm;
    float bias0 = (col0 < HH) ? bq1[col0] : 0.f;
    float bias1 = (has1 && col1 < HH) ? bq1[col1] : 0.f;
    bool v0 = col0 < HH, v1 = col1 < HH;
#pragma unroll
    for (int v = 0; v < 8; ++v) {
        int mm = v + g * 8;
        size_t row = (size_t)mt * 16 + mm;
        embq[row * HP + col0] = v0 ? (acc0[v] + bias0) : 0.f;
        if (has1) embq[row * HP + col1] = v1 ? (acc1[v] + bias1) : 0.f;
    }
}

// ------------------------------------------------- persistent rollout: 64 blocks x 16 rows, loop t
// 8 waves/block: halves the dependent-WMMA chain per stage vs 4 waves.
__global__ __launch_bounds__(256) void k_rollout(
    const float* __restrict__ actri, const float* __restrict__ embq,
    const float* __restrict__ pstoch, const float* __restrict__ pdeter,
    const float* __restrict__ n_pr, const float* __restrict__ n_po,
    const unsigned char* __restrict__ wsb,
    const float* __restrict__ bih, const float* __restrict__ bhh,
    const float* __restrict__ bp1, const float* __restrict__ bp2,
    const float* __restrict__ bq2, float* __restrict__ out) {
    __shared__ unsigned short stoch_bf[16 * 32];
    __shared__ unsigned short x_bf[16 * KD];
    __shared__ unsigned short det_bf[16 * KD];
    __shared__ unsigned short p1_bf[16 * KD];
    __shared__ unsigned short q1_bf[16 * KD];
    __shared__ float det_f[16 * HP];
    __shared__ float ms[2][16 * 64];

    const int tid = threadIdx.x, w = tid >> 5, L = tid & 31, g = L >> 4, lm = L & 15;
    const long long rowBase = (long long)blockIdx.x * 16;

    const unsigned* pWriS = (const unsigned*)(wsb + OFF_WRI);
    const unsigned* pWihR = (const unsigned*)(wsb + OFF_WIHR);
    const unsigned* pWihZ = (const unsigned*)(wsb + OFF_WIHZ);
    const unsigned* pWihN = (const unsigned*)(wsb + OFF_WIHN);
    const unsigned* pWhhR = (const unsigned*)(wsb + OFF_WHHR);
    const unsigned* pWhhZ = (const unsigned*)(wsb + OFF_WHHZ);
    const unsigned* pWhhN = (const unsigned*)(wsb + OFF_WHHN);
    const unsigned* pWp1  = (const unsigned*)(wsb + OFF_WP1);
    const unsigned* pWp2  = (const unsigned*)(wsb + OFF_WP2);
    const unsigned* pWq1d = (const unsigned*)(wsb + OFF_WQ1D);
    const unsigned* pWq2  = (const unsigned*)(wsb + OFF_WQ2);

    const size_t TB = (size_t)TT * BB;
    float* o_pm  = out;
    float* o_ps  = out + TB * SS;
    float* o_pst = out + 2 * TB * SS;
    float* o_d1  = out + 3 * TB * SS;
    float* o_qm  = o_d1 + TB * DD;
    float* o_qs  = o_qm + TB * SS;
    float* o_qst = o_qs + TB * SS;
    float* o_d2  = o_qst + TB * SS;

    // ---- init carries + zero pads
    for (int idx = tid; idx < 16 * 32; idx += 256) {
        int mm = idx >> 5, c = idx & 31;
        stoch_bf[idx] = (c < SS) ? f2bf(pstoch[(rowBase + mm) * SS + c]) : (unsigned short)0;
    }
    for (int idx = tid; idx < 16 * KD; idx += 256) {
        int mm = idx / KD, c = idx % KD;
        float dv = (c < DD) ? pdeter[(rowBase + mm) * DD + c] : 0.f;
        det_bf[idx] = f2bf(dv);
        x_bf[idx] = 0; p1_bf[idx] = 0; q1_bf[idx] = 0;
    }
    for (int idx = tid; idx < 16 * HP; idx += 256) {
        int mm = idx / HP, c = idx % HP;
        det_f[idx] = (c < DD) ? pdeter[(rowBase + mm) * DD + c] : 0.f;
    }
    __syncthreads();

    for (int t = 0; t < TT; ++t) {
        // ---- stage 1: x = elu(stoch@WriS + actri)
        for (int j = w; j < 13; j += 8) {
            f32x8 acc = {0.f, 0.f, 0.f, 0.f, 0.f, 0.f, 0.f, 0.f};
            Frag a, b;
            loadA(stoch_bf, 32, lm, g, 0, a);
            loadB(pWriS, j, L, b);
            acc = wmma_bf16(a, b, acc);
            int col = j * 16 + lm;
            bool cv = col < HH;
#pragma unroll
            for (int v = 0; v < 8; ++v) {
                int mm = v + g * 8;
                size_t tr = (size_t)t * BB + rowBase + mm;
                float val = acc[v] + actri[tr * HP + col];
                x_bf[mm * KD + col] = f2bf(cv ? elu1(val) : 0.f);
            }
        }
        __syncthreads();

        // ---- stage 2: GRU gates + deter update
        float dnew[2][8];
        for (int j = w; j < 13; j += 8) {
            int slot = (j - w) >> 3;
            f32x8 ra = {0.f,0.f,0.f,0.f,0.f,0.f,0.f,0.f}, za = ra, na = ra, ha = ra;
            for (int kt = 0; kt < 7; ++kt) {
                Frag ax, ad, b;
                loadA(x_bf, KD, lm, g, kt * 32, ax);
                loadA(det_bf, KD, lm, g, kt * 32, ad);
                loadB(pWihR, j * 7 + kt, L, b); ra = wmma_bf16(ax, b, ra);
                loadB(pWhhR, j * 7 + kt, L, b); ra = wmma_bf16(ad, b, ra);
                loadB(pWihZ, j * 7 + kt, L, b); za = wmma_bf16(ax, b, za);
                loadB(pWhhZ, j * 7 + kt, L, b); za = wmma_bf16(ad, b, za);
                loadB(pWihN, j * 7 + kt, L, b); na = wmma_bf16(ax, b, na);
                loadB(pWhhN, j * 7 + kt, L, b); ha = wmma_bf16(ad, b, ha);
            }
            int col = j * 16 + lm;
            bool valid = col < DD;
            int cc = valid ? col : 0;
            // column-invariant biases hoisted
            float rb = bih[cc] + bhh[cc];
            float zb = bih[DD + cc] + bhh[DD + cc];
            float nbx = bih[2 * DD + cc];
            float nbh = bhh[2 * DD + cc];
#pragma unroll
            for (int v = 0; v < 8; ++v) {
                int mm = v + g * 8;
                size_t tr = (size_t)t * BB + rowBase + mm;
                float r  = sigm(ra[v] + rb);
                float z  = sigm(za[v] + zb);
                float nn = tanhf(na[v] + nbx + r * (ha[v] + nbh));
                float dold = det_f[mm * HP + col];
                float dn = valid ? ((1.f - z) * nn + z * dold) : 0.f;
                dnew[slot][v] = dn;
                if (valid) { o_d1[tr * DD + col] = dn; o_d2[tr * DD + col] = dn; }
            }
        }
        __syncthreads();
        for (int j = w; j < 13; j += 8) {
            int slot = (j - w) >> 3, col = j * 16 + lm;
#pragma unroll
            for (int v = 0; v < 8; ++v) {
                int mm = v + g * 8;
                det_f[mm * HP + col] = dnew[slot][v];
                det_bf[mm * KD + col] = f2bf(dnew[slot][v]);
            }
        }
        __syncthreads();

        // ---- stage 3: p1 = elu(deter@Wp1+bp1) ; q1 = elu(deter@Wq1d + embq)
        for (int jj = w; jj < 26; jj += 8) {
            int isq = jj >= 13;
            int j = isq ? jj - 13 : jj;
            f32x8 acc = {0.f, 0.f, 0.f, 0.f, 0.f, 0.f, 0.f, 0.f};
            const unsigned* wb = isq ? pWq1d : pWp1;
            for (int kt = 0; kt < 7; ++kt) {
                Frag a, b;
                loadA(det_bf, KD, lm, g, kt * 32, a);
                loadB(wb, j * 7 + kt, L, b);
                acc = wmma_bf16(a, b, acc);
            }
            int col = j * 16 + lm;
            bool cv = col < HH;
            float pb = (!isq && cv) ? bp1[col] : 0.f;   // hoisted (embq already has bq1)
#pragma unroll
            for (int v = 0; v < 8; ++v) {
                int mm = v + g * 8;
                size_t tr = (size_t)t * BB + rowBase + mm;
                float add = isq ? embq[tr * HP + col] : pb;
                float val = acc[v] + add;
                unsigned short bv = f2bf(cv ? elu1(val) : 0.f);
                if (isq) q1_bf[mm * KD + col] = bv; else p1_bf[mm * KD + col] = bv;
            }
        }
        __syncthreads();

        // ---- stage 4a: prior/posterior heads -> mean / std
        for (int jj = w; jj < 8; jj += 8) {
            int isq = jj >> 2, j = jj & 3;
            f32x8 acc = {0.f, 0.f, 0.f, 0.f, 0.f, 0.f, 0.f, 0.f};
            const unsigned short* abuf = isq ? q1_bf : p1_bf;
            const unsigned* wb = isq ? pWq2 : pWp2;
            for (int kt = 0; kt < 7; ++kt) {
                Frag a, b;
                loadA(abuf, KD, lm, g, kt * 32, a);
                loadB(wb, j * 7 + kt, L, b);
                acc = wmma_bf16(a, b, acc);
            }
            int col = j * 16 + lm;
            bool inr = col < 2 * SS;
            bool ismean = col < SS;
            float bias = inr ? (isq ? bq2[col] : bp2[col]) : 0.f;  // hoisted
#pragma unroll
            for (int v = 0; v < 8; ++v) {
                int mm = v + g * 8;
                size_t tr = (size_t)t * BB + rowBase + mm;
                if (inr) {
                    float val = acc[v] + bias;
                    if (ismean) {
                        ms[isq][mm * 64 + col] = val;
                        (isq ? o_qm : o_pm)[tr * SS + col] = val;
                    } else {
                        float sd = softp01(val);
                        ms[isq][mm * 64 + col] = sd;
                        (isq ? o_qs : o_ps)[tr * SS + (col - SS)] = sd;
                    }
                }
            }
        }
        __syncthreads();
        // ---- stage 4b: stoch = mean + std * noise ; feed q_stoch to next step
        for (int idx = tid; idx < 2 * 16 * SS; idx += 256) {
            int isq = idx >= 16 * SS;
            int r = isq ? idx - 16 * SS : idx;
            int mm = r / SS, c = r % SS;
            size_t tr = (size_t)t * BB + rowBase + mm;
            float st = ms[isq][mm * 64 + c] +
                       ms[isq][mm * 64 + SS + c] * (isq ? n_po : n_pr)[tr * SS + c];
            (isq ? o_qst : o_pst)[tr * SS + c] = st;
            if (isq) stoch_bf[mm * 32 + c] = f2bf(st);
        }
        __syncthreads();
    }
}

// ---------------------------------------------------------------- launcher
extern "C" void kernel_launch(void* const* d_in, const int* in_sizes, int n_in,
                              void* d_out, int out_size, void* d_ws, size_t ws_size,
                              hipStream_t stream) {
    (void)in_sizes; (void)n_in; (void)out_size; (void)ws_size;
    const float* obs = (const float*)d_in[1];
    const float* act = (const float*)d_in[2];
    const float* pst = (const float*)d_in[3];
    const float* pdt = (const float*)d_in[4];
    const float* npr = (const float*)d_in[5];
    const float* npo = (const float*)d_in[6];
    const float* Wri = (const float*)d_in[7];
    const float* bri = (const float*)d_in[8];
    const float* Wih = (const float*)d_in[9];
    const float* bih = (const float*)d_in[10];
    const float* Whh = (const float*)d_in[11];
    const float* bhh = (const float*)d_in[12];
    const float* Wp1 = (const float*)d_in[13];
    const float* bp1 = (const float*)d_in[14];
    const float* Wp2 = (const float*)d_in[15];
    const float* bp2 = (const float*)d_in[16];
    const float* Wq1 = (const float*)d_in[17];
    const float* bq1 = (const float*)d_in[18];
    const float* Wq2 = (const float*)d_in[19];
    const float* bq2 = (const float*)d_in[20];
    unsigned char* wsb = (unsigned char*)d_ws;
    float* out = (float*)d_out;

    // pack all weights to bf16 WMMA fragment tiles (L2-resident, ~1.2 MB)
    k_pack<<<dim3(1, 13), 32, 0, stream>>>(Wri, HH, AA, 0,      SS, HH,    1, (unsigned short*)(wsb + OFF_WRI));
    k_pack<<<dim3(7, 13), 32, 0, stream>>>(Wih, 3 * DD, 0, 0,     HH, DD,  7, (unsigned short*)(wsb + OFF_WIHR));
    k_pack<<<dim3(7, 13), 32, 0, stream>>>(Wih, 3 * DD, 0, DD,    HH, DD,  7, (unsigned short*)(wsb + OFF_WIHZ));
    k_pack<<<dim3(7, 13), 32, 0, stream>>>(Wih, 3 * DD, 0, 2*DD,  HH, DD,  7, (unsigned short*)(wsb + OFF_WIHN));
    k_pack<<<dim3(7, 13), 32, 0, stream>>>(Whh, 3 * DD, 0, 0,     DD, DD,  7, (unsigned short*)(wsb + OFF_WHHR));
    k_pack<<<dim3(7, 13), 32, 0, stream>>>(Whh, 3 * DD, 0, DD,    DD, DD,  7, (unsigned short*)(wsb + OFF_WHHZ));
    k_pack<<<dim3(7, 13), 32, 0, stream>>>(Whh, 3 * DD, 0, 2*DD,  DD, DD,  7, (unsigned short*)(wsb + OFF_WHHN));
    k_pack<<<dim3(7, 13), 32, 0, stream>>>(Wp1, HH, 0, 0,         HH, HH,  7, (unsigned short*)(wsb + OFF_WP1));
    k_pack<<<dim3(7,  4), 32, 0, stream>>>(Wp2, 2 * SS, 0, 0,     HH, 2*SS,7, (unsigned short*)(wsb + OFF_WP2));
    k_pack<<<dim3(7, 13), 32, 0, stream>>>(Wq1, HH, 0, 0,         DD, HH,  7, (unsigned short*)(wsb + OFF_WQ1D));
    k_pack<<<dim3(32,13), 32, 0, stream>>>(Wq1, HH, DD, 0,        EE, HH, 32, (unsigned short*)(wsb + OFF_WQ1E));
    k_pack<<<dim3(7,  4), 32, 0, stream>>>(Wq2, 2 * SS, 0, 0,     HH, 2*SS,7, (unsigned short*)(wsb + OFF_WQ2));

    // hoisted, fully parallel GEMMs (removes 37% of FLOPs from the serial chain)
    long long actriTot = (long long)TT * BB * HP;
    k_actri<<<(unsigned)((actriTot + 255) / 256), 256, 0, stream>>>(
        act, Wri, bri, (float*)(wsb + OFF_ACTRI));
    k_embq<<<dim3(TT * BB / 16, 7), 32, 0, stream>>>(
        obs, (const unsigned*)(wsb + OFF_WQ1E), bq1, (float*)(wsb + OFF_EMBQ));

    // persistent recurrence: one block per 16 batch rows (8 waves), loops all T steps
    k_rollout<<<BB / 16, 256, 0, stream>>>(
        (const float*)(wsb + OFF_ACTRI), (const float*)(wsb + OFF_EMBQ),
        pst, pdt, npr, npo, wsb, bih, bhh, bp1, bp2, bq2, out);
}
